// HubbardPhaseGraphNet_79319456022873
// MI455X (gfx1250) — matrix operand order repaired and checked by reading, not compiled, not run on hardware
//
#include <hip/hip_runtime.h>
#include <hip/hip_bf16.h>
#include <math.h>

#define B_  8
#define N_  512
#define H_  64
#define L_  3
#define NODE_IN_ 5
#define G_IN_ 3
#define G_EMB_ 8
#define NCLS_ 4
#define LN_EPS_ 1e-5f

typedef __attribute__((ext_vector_type(16))) __bf16 v16bf;
typedef __attribute__((ext_vector_type(8)))  float  v8f;

union FragBF { v16bf v; unsigned u32[8]; unsigned short u16[16]; };

__device__ __forceinline__ unsigned short f2bf(float f) {
  union { float f; unsigned u; } in; in.f = f;
  unsigned u = in.u;
  unsigned r = u + 0x7FFFu + ((u >> 16) & 1u);   // round-to-nearest-even
  return (unsigned short)(r >> 16);
}

__device__ __forceinline__ float geluf(float x) {
  return 0.5f * x * (1.f + erff(x * 0.7071067811865475f));
}

// ---------------------------------------------------------------------------
// Pack u1/u2 weights (f32 -> bf16) into WMMA B-fragment order:
// packed[((kk*4 + nt)*32 + lane)*16 + i] = W[kk*32 + i + (lane>>4)*16][nt*16 + (lane&15)]
// so each wave reads its v16bf fragment as 32 contiguous bytes.
// ---------------------------------------------------------------------------
__global__ void pack_weights(const float* __restrict__ u1w,
                             const float* __restrict__ u2w,
                             unsigned short* __restrict__ u1p,
                             unsigned short* __restrict__ u2p) {
  int idx = blockIdx.x * blockDim.x + threadIdx.x;
  const int U1 = L_ * 128 * 64;
  const int U2 = L_ * 64 * 64;
  if (idx < U1) {
    int l = idx / 8192, r = idx % 8192;
    int i = r & 15, lane = (r >> 4) & 31, nt = (r >> 9) & 3, kk = (r >> 11) & 3;
    int k = kk * 32 + i + ((lane >> 4) & 1) * 16;
    int n = nt * 16 + (lane & 15);
    u1p[idx] = f2bf(u1w[l * 8192 + k * 64 + n]);
  } else if (idx < U1 + U2) {
    int j = idx - U1;
    int l = j / 4096, r = j % 4096;
    int i = r & 15, lane = (r >> 4) & 31, nt = (r >> 9) & 3, kk = (r >> 11) & 1;
    int k = kk * 32 + i + ((lane >> 4) & 1) * 16;
    int n = nt * 16 + (lane & 15);
    u2p[j] = f2bf(u2w[l * 4096 + k * 64 + n]);
  }
}

// ---------------------------------------------------------------------------
// h0 = concat(nodes, gp(global)) @ np_w + np_b, masked. One block per (b,n).
// ---------------------------------------------------------------------------
__global__ void embed_kernel(const float* __restrict__ nodes,
                             const float* __restrict__ gf,
                             const float* __restrict__ gpw,
                             const float* __restrict__ gpb,
                             const float* __restrict__ npw,
                             const float* __restrict__ npb,
                             const float* __restrict__ mask,
                             float* __restrict__ h) {
  int bi = blockIdx.x;          // b*N + n
  int b  = bi >> 9;
  int t  = threadIdx.x;         // output channel
  float acc = npb[t];
  const float* nd = nodes + (size_t)bi * NODE_IN_;
  #pragma unroll
  for (int c = 0; c < NODE_IN_; ++c) acc += nd[c] * npw[c * H_ + t];
  #pragma unroll
  for (int e = 0; e < G_EMB_; ++e) {
    float ge = gpb[e];
    #pragma unroll
    for (int c = 0; c < G_IN_; ++c) ge += gf[b * G_IN_ + c] * gpw[c * G_EMB_ + e];
    acc += ge * npw[(NODE_IN_ + e) * H_ + t];
  }
  h[(size_t)bi * H_ + t] = acc * mask[bi];
}

// ---------------------------------------------------------------------------
// deg[b,i] = max(sum_j adj, 1). One wave32 per row.
// ---------------------------------------------------------------------------
__global__ void deg_kernel(const float* __restrict__ adj, float* __restrict__ deg) {
  int t = threadIdx.x;
  int lane = t & 31;
  int r = blockIdx.x * 8 + (t >> 5);
  const float* arow = adj + (size_t)r * N_;
  float s = 0.f;
  for (int j = lane; j < N_; j += 32) s += arow[j];
  #pragma unroll
  for (int m = 16; m >= 1; m >>= 1) s += __shfl_xor(s, m, 32);
  if (lane == 0) deg[r] = fmaxf(s, 1.f);
}

// ---------------------------------------------------------------------------
// msg[b,i,h] = sum_j sigmoid(e*gw_h+gb_h)*adj * h[b,j,h] / deg.
// One 256-thread block per (b,i): 64 channels x 4 j-chunks, LDS reduce.
// adj/e reads broadcast within each 64-thread group; h reads coalesced.
// ---------------------------------------------------------------------------
__global__ void msg_kernel(const float* __restrict__ adj,
                           const float* __restrict__ ea,
                           const float* __restrict__ gw,
                           const float* __restrict__ gb,
                           const float* __restrict__ h,
                           const float* __restrict__ deg,
                           float* __restrict__ msg) {
  int bi = blockIdx.x;
  int b  = bi >> 9;
  int t  = threadIdx.x;
  int hc = t & 63, jc = t >> 6;
  float gwh = gw[hc], gbh = gb[hc];
  const float* arow = adj + (size_t)bi * N_;
  const float* erow = ea  + (size_t)bi * N_;
  const float* hb   = h   + (size_t)b * N_ * H_;
  float acc = 0.f;
  int j0 = jc * 128;
  for (int k = 0; k < 128; ++k) {
    int j = j0 + k;
    float a = arow[j];
    float x = erow[j] * gwh + gbh;
    float g = a / (1.f + __expf(-x));
    acc += g * hb[(size_t)j * H_ + hc];
  }
  __shared__ float red[256];
  red[t] = acc;
  __syncthreads();
  if (jc == 0) {
    float s = acc + red[hc + 64] + red[hc + 128] + red[hc + 192];
    msg[(size_t)bi * H_ + hc] = s / deg[bi];
  }
}

// ---------------------------------------------------------------------------
// WMMA update: per 16-row strip (4 waves):
//   z = [h | msg] (bf16 in LDS) -> 4x wmma_bf16 (K=128) -> +b1, gelu (bf16 LDS)
//   -> 2x wmma_bf16 (K=64) -> +b2, residual*mask, LayerNorm -> h (in place).
// ---------------------------------------------------------------------------
__global__ void update_kernel(float* __restrict__ h,
                              const float* __restrict__ msg,
                              const unsigned short* __restrict__ u1p,
                              const unsigned short* __restrict__ u2p,
                              const float* __restrict__ u1b,
                              const float* __restrict__ u2b,
                              const float* __restrict__ lng,
                              const float* __restrict__ lnb,
                              const float* __restrict__ mask) {
  __shared__ __align__(16) unsigned short zt[16 * 128];  // z tile bf16
  __shared__ __align__(16) unsigned short a1[16 * 64];   // gelu act bf16
  __shared__ __align__(16) float accf[16 * 64];          // upd f32

  int t = threadIdx.x;
  int R = blockIdx.x * 16;

  // Stage z = [h | msg] as bf16
  for (int idx = t; idx < 16 * 128; idx += 128) {
    int m = idx >> 7, c = idx & 127;
    int r = R + m;
    float v = (c < H_) ? h[(size_t)r * H_ + c] : msg[(size_t)r * H_ + (c - H_)];
    zt[idx] = f2bf(v);
  }
  __syncthreads();

  int wave = t >> 5, lane = t & 31;
  int mrow = lane & 15;
  int hi   = (lane >> 4) & 1;

  // GEMM1: [16x128] @ [128x16] per wave, K in 4 steps of 32
  v8f c1 = {};
  #pragma unroll
  for (int kk = 0; kk < 4; ++kk) {
    FragBF af, bf;
    #pragma unroll
    for (int p = 0; p < 8; ++p) {
      int i0 = 2 * p;
      int k = kk * 32 + hi * 16 + i0;  // A 16x32 bf16 layout (contiguous k = 16h + s)
      af.u32[p] = *(const unsigned*)&zt[mrow * 128 + k];
    }
    bf.v = *(const v16bf*)(u1p + ((kk * 4 + wave) * 32 + lane) * 16);
    c1 = __builtin_amdgcn_wmma_f32_16x16x32_bf16(false, af.v, false, bf.v,
                                                 (short)0, c1, false, false);
  }
  #pragma unroll
  for (int v = 0; v < 8; ++v) {
    int m  = v + hi * 8;
    int nn = wave * 16 + (lane & 15);
    a1[m * 64 + nn] = f2bf(geluf(c1[v] + u1b[nn]));
  }
  __syncthreads();

  // GEMM2: [16x64] @ [64x16] per wave, K in 2 steps of 32
  v8f c2 = {};
  #pragma unroll
  for (int kk = 0; kk < 2; ++kk) {
    FragBF af, bf;
    #pragma unroll
    for (int p = 0; p < 8; ++p) {
      int i0 = 2 * p;
      int k = kk * 32 + hi * 16 + i0;
      af.u32[p] = *(const unsigned*)&a1[mrow * 64 + k];
    }
    bf.v = *(const v16bf*)(u2p + ((kk * 4 + wave) * 32 + lane) * 16);
    c2 = __builtin_amdgcn_wmma_f32_16x16x32_bf16(false, af.v, false, bf.v,
                                                 (short)0, c2, false, false);
  }
  #pragma unroll
  for (int v = 0; v < 8; ++v) {
    int m  = v + hi * 8;
    int nn = wave * 16 + (lane & 15);
    accf[m * 64 + nn] = c2[v] + u2b[nn];
  }
  __syncthreads();

  // Residual + LayerNorm, one thread per row
  if (t < 16) {
    int r = R + t;
    float mk = mask[r];
    float x[64];
    float mu = 0.f;
    #pragma unroll
    for (int c = 0; c < 64; ++c) { x[c] = h[(size_t)r * 64 + c] + accf[t * 64 + c] * mk; mu += x[c]; }
    mu *= (1.f / 64.f);
    float var = 0.f;
    #pragma unroll
    for (int c = 0; c < 64; ++c) { float d = x[c] - mu; var += d * d; }
    var *= (1.f / 64.f);
    float rs = rsqrtf(var + LN_EPS_);
    #pragma unroll
    for (int c = 0; c < 64; ++c)
      h[(size_t)r * 64 + c] = (x[c] - mu) * rs * lng[c] + lnb[c];
  }
}

// ---------------------------------------------------------------------------
// Mean pool + 2-layer head. One block per batch, one thread per channel.
// ---------------------------------------------------------------------------
__global__ void head_kernel(const float* __restrict__ h,
                            const float* __restrict__ mask,
                            const float* __restrict__ h1w,
                            const float* __restrict__ h1b,
                            const float* __restrict__ h2w,
                            const float* __restrict__ h2b,
                            float* __restrict__ out) {
  int b = blockIdx.x, t = threadIdx.x;
  const float* hb = h + (size_t)b * N_ * H_;
  const float* mb = mask + (size_t)b * N_;
  float s = 0.f, mc = 0.f;
  for (int n = 0; n < N_; ++n) { float mk = mb[n]; s += hb[(size_t)n * H_ + t] * mk; mc += mk; }
  float pooled = s / fmaxf(mc, 1.f);
  __shared__ float pl[64], q[64];
  pl[t] = pooled;
  __syncthreads();
  float a = h1b[t];
  for (int k = 0; k < 64; ++k) a += pl[k] * h1w[k * 64 + t];
  q[t] = geluf(a);
  __syncthreads();
  if (t < NCLS_) {
    float o = h2b[t];
    for (int k = 0; k < 64; ++k) o += q[k] * h2w[k * NCLS_ + t];
    out[b * NCLS_ + t] = o;
  }
}

extern "C" void kernel_launch(void* const* d_in, const int* in_sizes, int n_in,
                              void* d_out, int out_size, void* d_ws, size_t ws_size,
                              hipStream_t stream) {
  const float* nodes = (const float*)d_in[0];
  const float* adj   = (const float*)d_in[1];
  const float* ea    = (const float*)d_in[2];
  const float* mask  = (const float*)d_in[3];
  const float* gf    = (const float*)d_in[4];
  const float* gpw   = (const float*)d_in[5];
  const float* gpb   = (const float*)d_in[6];
  const float* npw   = (const float*)d_in[7];
  const float* npb   = (const float*)d_in[8];
  const float* gatew = (const float*)d_in[9];
  const float* gateb = (const float*)d_in[10];
  const float* u1w   = (const float*)d_in[11];
  const float* u1b   = (const float*)d_in[12];
  const float* u2w   = (const float*)d_in[13];
  const float* u2b   = (const float*)d_in[14];
  const float* lng   = (const float*)d_in[15];
  const float* lnb   = (const float*)d_in[16];
  const float* h1w   = (const float*)d_in[17];
  const float* h1b   = (const float*)d_in[18];
  const float* h2w   = (const float*)d_in[19];
  const float* h2b   = (const float*)d_in[20];
  float* out = (float*)d_out;

  char* ws = (char*)d_ws;
  size_t off = 0;
  float* h   = (float*)(ws + off); off += (size_t)B_ * N_ * H_ * 4;
  float* msg = (float*)(ws + off); off += (size_t)B_ * N_ * H_ * 4;
  float* deg = (float*)(ws + off); off += ((size_t)B_ * N_ * 4 + 255) & ~(size_t)255;
  unsigned short* u1p = (unsigned short*)(ws + off); off += (size_t)L_ * 128 * 64 * 2;
  unsigned short* u2p = (unsigned short*)(ws + off); off += (size_t)L_ * 64 * 64 * 2;

  pack_weights<<<(L_ * 128 * 64 + L_ * 64 * 64 + 255) / 256, 256, 0, stream>>>(u1w, u2w, u1p, u2p);
  embed_kernel<<<B_ * N_, 64, 0, stream>>>(nodes, gf, gpw, gpb, npw, npb, mask, h);
  deg_kernel<<<B_ * N_ / 8, 256, 0, stream>>>(adj, deg);
  for (int l = 0; l < L_; ++l) {
    msg_kernel<<<B_ * N_, 256, 0, stream>>>(adj, ea, gatew + l * H_, gateb + l * H_, h, deg, msg);
    update_kernel<<<B_ * N_ / 16, 128, 0, stream>>>(h, msg,
        u1p + (size_t)l * 128 * 64, u2p + (size_t)l * 64 * 64,
        u1b + l * H_, u2b + l * H_, lng + l * H_, lnb + l * H_, mask);
  }
  head_kernel<<<B_, 64, 0, stream>>>(h, mask, h1w, h1b, h2w, h2b, out);
}
